// EnhancedGCN_3590592660348
// MI455X (gfx1250) — matrix operand order repaired and checked by reading, not compiled, not run on hardware
//
#include <hip/hip_runtime.h>
#include <math.h>

// ---------------- types for WMMA ----------------
typedef __attribute__((ext_vector_type(16))) __bf16 v16bf;
typedef __attribute__((ext_vector_type(8)))  float  v8f;

__device__ __forceinline__ unsigned short f2bf(float f) {
    unsigned int u = __float_as_uint(f);
    unsigned int r = (u + 0x7FFFu + ((u >> 16) & 1u)) >> 16;   // RNE
    return (unsigned short)r;
}

__device__ __forceinline__ float leaky02(float x) { return x > 0.f ? x : 0.2f * x; }

__device__ __forceinline__ void atomicMaxF(float* addr, float val) {
    unsigned int* ua = (unsigned int*)addr;
    unsigned int old = *ua;
    while (__uint_as_float(old) < val) {
        unsigned int assumed = old;
        old = atomicCAS(ua, assumed, __float_as_uint(val));
        if (old == assumed) break;
    }
}

// ---------------- conversion kernels ----------------
__global__ void k_f32_to_bf16(const float* __restrict__ S, unsigned short* __restrict__ D, int n) {
    int t = blockIdx.x * 256 + threadIdx.x;
    if (t < n) D[t] = f2bf(S[t]);
}

// S is R x C row-major f32; D is C x R row-major bf16 (transpose+convert)
__global__ void k_transpose_bf16(const float* __restrict__ S, unsigned short* __restrict__ D, int R, int C) {
    int t = blockIdx.x * 256 + threadIdx.x;
    if (t >= R * C) return;
    int c = t / R, r = t - c * R;
    D[t] = f2bf(S[r * C + c]);
}

// ---------------- WMMA bf16 GEMM:  C[M x Nout] = A[M x K] * BT[Nout x K]^T ----
// block = 256 threads = 8 waves; each wave owns a 16 x Nout stripe of C.
// Whole BT (Nout x K bf16, <= 64KB) is staged in LDS once per block: all B
// fragment reads become ds_load_b128, only the A stream touches global memory.
template <int NT>  // NT = Nout/16 (compile-time so accumulators stay in VGPRs)
__global__ void k_gemm_bf16(const unsigned short* __restrict__ A,
                            const unsigned short* __restrict__ BT,
                            float* __restrict__ C,
                            int M, int K, int Nout) {
    extern __shared__ unsigned short lds_bt[];   // Nout*K bf16

    // cooperative stage: 16B per thread per step
    const int nvec = (Nout * K) >> 3;            // # of uint4 chunks
    for (int i = threadIdx.x; i < nvec; i += 256)
        ((uint4*)lds_bt)[i] = ((const uint4*)BT)[i];
    __syncthreads();

    const int lane = threadIdx.x & 31;
    const int wave = threadIdx.x >> 5;
    const int row0 = (blockIdx.x * 8 + wave) * 16;
    if (row0 >= M) return;                       // wave-uniform exit: EXEC stays all-ones

    const int r     = lane & 15;                 // row (A) / column (B) within tile
    const int hi    = lane >> 4;                 // half-wave select
    const int akoff = hi * 8;                    // A: K groups {0..7,16..23} / {8..15,24..31}
    const int bkoff = hi * 8;                    // B: K groups {0..7,16..23} / {8..15,24..31}

    int arow = row0 + r; if (arow >= M) arow = M - 1;   // clamp keeps EXEC full
    const unsigned short* pA = A + (size_t)arow * K;
    const unsigned short* pBl = lds_bt + (size_t)r * K + bkoff;   // col = t*16 + r

    v8f acc[NT];
#pragma unroll
    for (int t = 0; t < NT; ++t)
#pragma unroll
        for (int i = 0; i < 8; ++i) acc[t][i] = 0.0f;

    for (int kk = 0; kk < K; kk += 32) {
        if (kk + 32 < K) __builtin_prefetch(pA + kk + 32, 0, 3);   // near-scope prefetch of A stream

        union { uint4 u[2]; v16bf v; } af;
        af.u[0] = *(const uint4*)(pA + kk + akoff);        // K = kk+akoff    .. +7
        af.u[1] = *(const uint4*)(pA + kk + akoff + 16);   // K = kk+akoff+16 .. +23

#pragma unroll
        for (int t = 0; t < NT; ++t) {
            const unsigned short* pB = pBl + (size_t)t * 16 * K + kk;
            union { uint4 u[2]; v16bf v; } bf_;
            bf_.u[0] = *(const uint4*)(pB);                // K = kk+bkoff    .. +7
            bf_.u[1] = *(const uint4*)(pB + 16);           // K = kk+bkoff+16 .. +23
            acc[t] = __builtin_amdgcn_wmma_f32_16x16x32_bf16(
                false, af.v, false, bf_.v, (short)0, acc[t], false, false);
        }
    }

    const int mbase = row0 + hi * 8;   // C layout: lanes 0-15 -> M 0..7, lanes 16-31 -> M 8..15
#pragma unroll
    for (int t = 0; t < NT; ++t) {
        const int col = t * 16 + r;
#pragma unroll
        for (int v = 0; v < 8; ++v) {
            const int mr = mbase + v;
            if (mr < M) C[(size_t)mr * Nout + col] = acc[t][v];
        }
    }
}

// ---------------- GCN pipeline ----------------
__global__ void k_init_deg(float* deg, int N) {
    int t = blockIdx.x * 256 + threadIdx.x;
    if (t < N) deg[t] = 1.0f;                    // self loop
}

__global__ void k_deg_edges(const int* __restrict__ dst, float* deg, int E) {
    int t = blockIdx.x * 256 + threadIdx.x;
    if (t < E) atomicAdd(&deg[dst[t]], 1.0f);
}

__global__ void k_deg_to_dinv(float* deg, int N) {
    int t = blockIdx.x * 256 + threadIdx.x;
    if (t < N) deg[t] = rsqrtf(deg[t]);
}

// acc = self-loop contribution: xw * dinv^2
__global__ void k_gcn_init_acc(const float* __restrict__ xw, const float* __restrict__ dinv,
                               float* __restrict__ acc, int N) {
    int t = blockIdx.x * 256 + threadIdx.x;
    if (t >= N * 128) return;
    int n = t >> 7;
    float di = dinv[n];
    acc[t] = xw[t] * di * di;
}

// one wave per edge; 4 features per lane
__global__ void k_gcn_scatter(const int* __restrict__ src, const int* __restrict__ dst,
                              const float* __restrict__ dinv, const float* __restrict__ xw,
                              float* __restrict__ acc, int E) {
    int e = blockIdx.x * 8 + (threadIdx.x >> 5);
    if (e >= E) return;
    int lane = threadIdx.x & 31;
    int s = src[e], d = dst[e];
    float norm = dinv[s] * dinv[d];
    const float* xs = xw + (size_t)s * 128;
    float* ad = acc + (size_t)d * 128;
#pragma unroll
    for (int i = 0; i < 4; ++i) {
        int f = lane + i * 32;
        atomicAdd(ad + f, xs[f] * norm);
    }
}

// h = bf16(relu(acc + b1))
__global__ void k_gcn_finalize(const float* __restrict__ acc, const float* __restrict__ b1,
                               unsigned short* __restrict__ h, int N) {
    int t = blockIdx.x * 256 + threadIdx.x;
    if (t >= N * 128) return;
    float v = acc[t] + b1[t & 127];
    h[t] = f2bf(v > 0.f ? v : 0.f);
}

// ---------------- GAT pipeline ----------------
// per (node, head): attention scores + m init with self-edge value
__global__ void k_att_prep(const float* __restrict__ h2, const float* __restrict__ att_s,
                           const float* __restrict__ att_d, float* __restrict__ as_,
                           float* __restrict__ ad_, float* __restrict__ m, int N) {
    int t = blockIdx.x * 256 + threadIdx.x;
    if (t >= N * 3) return;
    int n = t / 3, hd = t - n * 3;
    const float* row = h2 + (size_t)n * 192 + hd * 64;
    const float* ws = att_s + hd * 64;
    const float* wd = att_d + hd * 64;
    float s = 0.f, d = 0.f;
#pragma unroll 8
    for (int c = 0; c < 64; ++c) { s += row[c] * ws[c]; d += row[c] * wd[c]; }
    as_[t] = s; ad_[t] = d;
    m[t] = leaky02(s + d);                       // self-edge score seeds the segment max
}

__global__ void k_edge_max(const int* __restrict__ src, const int* __restrict__ dst,
                           const float* __restrict__ as_, const float* __restrict__ ad_,
                           float* __restrict__ m, int E) {
    int t = blockIdx.x * 256 + threadIdx.x;
    if (t >= E * 3) return;
    int e = t / 3, hd = t - e * 3;
    int s = src[e], d = dst[e];
    float ev = leaky02(as_[s * 3 + hd] + ad_[d * 3 + hd]);
    atomicMaxF(&m[d * 3 + hd], ev);
}

__global__ void k_init_ssum(const float* __restrict__ as_, const float* __restrict__ ad_,
                            const float* __restrict__ m, float* __restrict__ ss, int N) {
    int t = blockIdx.x * 256 + threadIdx.x;
    if (t >= N * 3) return;
    float ev = leaky02(as_[t] + ad_[t]);
    ss[t] = __expf(ev - m[t]);                   // self-edge term
}

__global__ void k_edge_sum(const int* __restrict__ src, const int* __restrict__ dst,
                           const float* __restrict__ as_, const float* __restrict__ ad_,
                           const float* __restrict__ m, float* __restrict__ ss, int E) {
    int t = blockIdx.x * 256 + threadIdx.x;
    if (t >= E * 3) return;
    int e = t / 3, hd = t - e * 3;
    int s = src[e], d = dst[e];
    float ev = leaky02(as_[s * 3 + hd] + ad_[d * 3 + hd]);
    atomicAdd(&ss[d * 3 + hd], __expf(ev - m[d * 3 + hd]));
}

// out = self contribution + bias
__global__ void k_init_out(const float* __restrict__ h2, const float* __restrict__ as_,
                           const float* __restrict__ ad_, const float* __restrict__ m,
                           const float* __restrict__ ss, const float* __restrict__ b2,
                           float* __restrict__ out, int N) {
    int t = blockIdx.x * 256 + threadIdx.x;
    if (t >= N * 192) return;
    int n = t / 192, f = t - n * 192, hd = f >> 6;
    float ev = leaky02(as_[n * 3 + hd] + ad_[n * 3 + hd]);
    float alpha = __expf(ev - m[n * 3 + hd]) / ss[n * 3 + hd];
    out[t] = h2[t] * alpha + b2[f];
}

// one wave per edge; 6 features per lane
__global__ void k_edge_scatter(const int* __restrict__ src, const int* __restrict__ dst,
                               const float* __restrict__ as_, const float* __restrict__ ad_,
                               const float* __restrict__ m, const float* __restrict__ ss,
                               const float* __restrict__ h2, float* __restrict__ out, int E) {
    int e = blockIdx.x * 8 + (threadIdx.x >> 5);
    if (e >= E) return;
    int lane = threadIdx.x & 31;
    int s = src[e], d = dst[e];
    float al[3];
#pragma unroll
    for (int hd = 0; hd < 3; ++hd) {
        float ev = leaky02(as_[s * 3 + hd] + ad_[d * 3 + hd]);
        al[hd] = __expf(ev - m[d * 3 + hd]) / ss[d * 3 + hd];
    }
    const float* hs = h2 + (size_t)s * 192;
    float* od = out + (size_t)d * 192;
#pragma unroll
    for (int i = 0; i < 6; ++i) {
        int f = lane + i * 32;
        atomicAdd(od + f, hs[f] * al[f >> 6]);
    }
}

__global__ void k_relu(float* __restrict__ out, int n) {
    int t = blockIdx.x * 256 + threadIdx.x;
    if (t < n) { float v = out[t]; out[t] = v > 0.f ? v : 0.f; }
}

// ---------------- host launcher ----------------
static inline int cdiv(long long a, long long b) { return (int)((a + b - 1) / b); }

extern "C" void kernel_launch(void* const* d_in, const int* in_sizes, int n_in,
                              void* d_out, int out_size, void* d_ws, size_t ws_size,
                              hipStream_t stream) {
    const float* x       = (const float*)d_in[0];
    const int*   ei      = (const int*)  d_in[1];
    const float* W1      = (const float*)d_in[2];
    const float* b1      = (const float*)d_in[3];
    const float* W2      = (const float*)d_in[4];
    const float* att_src = (const float*)d_in[5];
    const float* att_dst = (const float*)d_in[6];
    const float* b2      = (const float*)d_in[7];
    float* out = (float*)d_out;

    const int IC = 256, HID = 128, OC = 192;
    const int N = in_sizes[0] / IC;
    const int E = in_sizes[1] / 2;
    const int* src = ei;
    const int* dst = ei + E;

    // workspace carve-out (256B aligned regions)
    char* ws = (char*)d_ws;
    size_t off = 0;
    auto take = [&](size_t bytes) -> char* {
        char* p = ws + off;
        off += (bytes + 255) & ~(size_t)255;
        return p;
    };
    unsigned short* xA  = (unsigned short*)take((size_t)N * IC * 2);   // bf16 x
    unsigned short* W1T = (unsigned short*)take((size_t)HID * IC * 2); // bf16 W1^T [128x256]
    unsigned short* W2T = (unsigned short*)take((size_t)OC * HID * 2); // bf16 W2^T [192x128]
    float* deg = (float*)take((size_t)N * 4);                          // deg -> dinv
    float* xw  = (float*)take((size_t)N * HID * 4);                    // x @ W1
    unsigned short* hbf = (unsigned short*)take((size_t)N * HID * 2);  // bf16 relu(gcn)
    float* as_ = (float*)take((size_t)N * 3 * 4);
    float* ad_ = (float*)take((size_t)N * 3 * 4);
    float* mX  = (float*)take((size_t)N * 3 * 4);
    float* sX  = (float*)take((size_t)N * 3 * 4);
    float* h2  = (float*)take((size_t)N * OC * 4);                     // h @ W2
    float* acc = h2;  // GCN accumulator aliases h2 (lifetimes disjoint)

    // ---- precision conversion / weight transposes ----
    k_f32_to_bf16<<<cdiv((long long)N * IC, 256), 256, 0, stream>>>(x, xA, N * IC);
    k_transpose_bf16<<<cdiv(IC * HID, 256), 256, 0, stream>>>(W1, W1T, IC, HID);
    k_transpose_bf16<<<cdiv(HID * OC, 256), 256, 0, stream>>>(W2, W2T, HID, OC);

    // ---- GCN ----
    k_init_deg<<<cdiv(N, 256), 256, 0, stream>>>(deg, N);
    k_deg_edges<<<cdiv(E, 256), 256, 0, stream>>>(dst, deg, E);
    k_deg_to_dinv<<<cdiv(N, 256), 256, 0, stream>>>(deg, N);
    k_gemm_bf16<8><<<cdiv(N, 128), 256, (size_t)HID * IC * 2, stream>>>(xA, W1T, xw, N, IC, HID);
    k_gcn_init_acc<<<cdiv((long long)N * HID, 256), 256, 0, stream>>>(xw, deg, acc, N);
    k_gcn_scatter<<<cdiv(E, 8), 256, 0, stream>>>(src, dst, deg, xw, acc, E);
    k_gcn_finalize<<<cdiv((long long)N * HID, 256), 256, 0, stream>>>(acc, b1, hbf, N);

    // ---- GAT ----
    k_gemm_bf16<12><<<cdiv(N, 128), 256, (size_t)OC * HID * 2, stream>>>(hbf, W2T, h2, N, HID, OC);
    k_att_prep<<<cdiv((long long)N * 3, 256), 256, 0, stream>>>(h2, att_src, att_dst, as_, ad_, mX, N);
    k_edge_max<<<cdiv((long long)E * 3, 256), 256, 0, stream>>>(src, dst, as_, ad_, mX, E);
    k_init_ssum<<<cdiv((long long)N * 3, 256), 256, 0, stream>>>(as_, ad_, mX, sX, N);
    k_edge_sum<<<cdiv((long long)E * 3, 256), 256, 0, stream>>>(src, dst, as_, ad_, mX, sX, E);
    k_init_out<<<cdiv((long long)N * OC, 256), 256, 0, stream>>>(h2, as_, ad_, mX, sX, b2, out, N);
    k_edge_scatter<<<cdiv(E, 8), 256, 0, stream>>>(src, dst, as_, ad_, mX, sX, h2, out, E);
    k_relu<<<cdiv((long long)N * OC, 256), 256, 0, stream>>>(out, N * OC);
}
